// Model_25074019074638
// MI455X (gfx1250) — matrix operand order repaired and checked by reading, not compiled, not run on hardware
//
#include <hip/hip_runtime.h>
#include <hip/hip_bf16.h>

typedef _Float16 f16;
typedef __attribute__((ext_vector_type(16))) _Float16 v16h;
typedef __attribute__((ext_vector_type(8)))  float    v8f;

#define LNUM   6
#define DMODEL 1024
#define NHEAD  16
#define HDIM   64
#define FFDIM  4096
#define VOCAB  1000
#define SEQ    1024
#define BATCH  2
#define MROWS  (SEQ * BATCH)   // 2048 token rows

union FragU { v16h v; unsigned u[8]; };

// A-matrix 16x32 f16 fragment from a row-major [16][ld] tile (LDS or global).
// Layout: lanes 0-15 row M=lane, K=0..15; lanes 16-31 same row, K=16..31.
__device__ __forceinline__ v16h frag_a(const f16* tile, int ld, int lane) {
  int row = lane & 15;
  int ko  = (lane >> 4) << 4;          // 0 or 16
  FragU r;
  const f16* p = tile + row * ld + ko;
#pragma unroll
  for (int i = 0; i < 4; ++i) {
    r.u[i]     = *(const unsigned*)(p + 2 * i);
    r.u[i + 4] = *(const unsigned*)(p + 8 + 2 * i);
  }
  return r.v;
}

// B-matrix 32x16 f16 fragment from row-major [16 n-rows][ld k-cols] tile
// (i.e. W stored (N,K)): lane holds col n=lane&15, VGPR v K={2v,2v+1}(+16 hi lanes)
__device__ __forceinline__ v16h frag_b(const f16* tile, int ld, int lane) {
  int col = lane & 15;
  int kb  = (lane >> 4) << 4;          // 0 or 16
  FragU r;
  const f16* p = tile + col * ld + kb;
#pragma unroll
  for (int i = 0; i < 8; ++i) r.u[i] = *(const unsigned*)(p + 2 * i);
  return r.v;
}

// ---------------------------------------------------------------- GEMM
// C(M,N) = act( A(M,K)f16 @ W(N,K)^T f16 + bias + residual )
// BLK 128x128, 8 waves (4x2), each wave = 32(M) x 64(N):
//   2 A-fragments x 4 B-fragments = 8 WMMAs per 32-deep K step.
template <bool RELU>
__global__ __launch_bounds__(256) void gemm_kernel(
    const f16* __restrict__ A, const f16* __restrict__ W,
    const float* __restrict__ bias, const float* __restrict__ residual,
    float* __restrict__ outf, f16* __restrict__ outh,
    int M, int N, int K) {
  __shared__ __align__(16) f16 As[128][40];
  __shared__ __align__(16) f16 Ws[128][40];
  int tid  = threadIdx.x;
  int lane = tid & 31;
  int wave = tid >> 5;
  int wm = wave >> 1, wn = wave & 1;            // 4x2 wave grid
  int bm = blockIdx.y * 128, bn = blockIdx.x * 128;
  int lrow = tid >> 1;                          // 0..127
  int lcol = (tid & 1) * 16;                    // 0 or 16

  v8f acc[2][4] = {};

  const f16* ga = A + (size_t)(bm + lrow) * K + lcol;
  int nrow = bn + lrow;
  const f16* gw = W + (size_t)nrow * K + lcol;

  for (int k0 = 0; k0 < K; k0 += 32) {
    // run the HBM->L2 stream one tile ahead of the LDS fill
    if (k0 + 32 < K) {
      __builtin_prefetch(ga + k0 + 32, 0, 1);
      if (nrow < N) __builtin_prefetch(gw + k0 + 32, 0, 1);
    }
    __syncthreads();
    {
      uint4 d0 = *(const uint4*)(ga + k0);
      uint4 d1 = *(const uint4*)(ga + k0 + 8);
      *(uint4*)&As[lrow][lcol]     = d0;
      *(uint4*)&As[lrow][lcol + 8] = d1;
    }
    {
      uint4 d0 = make_uint4(0u, 0u, 0u, 0u), d1 = make_uint4(0u, 0u, 0u, 0u);
      if (nrow < N) {
        d0 = *(const uint4*)(gw + k0);
        d1 = *(const uint4*)(gw + k0 + 8);
      }
      *(uint4*)&Ws[lrow][lcol]     = d0;
      *(uint4*)&Ws[lrow][lcol + 8] = d1;
    }
    __syncthreads();
    v16h a0 = frag_a(&As[wm * 32][0], 40, lane);
    v16h a1 = frag_a(&As[wm * 32 + 16][0], 40, lane);
#pragma unroll
    for (int j = 0; j < 4; ++j) {
      v16h b = frag_b(&Ws[wn * 64 + j * 16][0], 40, lane);
      acc[0][j] = __builtin_amdgcn_wmma_f32_16x16x32_f16(false, a0, false, b, (short)0, acc[0][j], false, false);
      acc[1][j] = __builtin_amdgcn_wmma_f32_16x16x32_f16(false, a1, false, b, (short)0, acc[1][j], false, false);
    }
  }

  int colb = lane & 15;
  int rowh = (lane >> 4) << 3;
#pragma unroll
  for (int i = 0; i < 2; ++i) {
    int mbase = bm + wm * 32 + i * 16 + rowh;
#pragma unroll
    for (int j = 0; j < 4; ++j) {
      int n = bn + wn * 64 + j * 16 + colb;
      if (n < N) {
        float bv = bias ? bias[n] : 0.f;
#pragma unroll
        for (int r = 0; r < 8; ++r) {
          int m = mbase + r;
          float val = acc[i][j][r] + bv;
          if (residual) val += residual[(size_t)m * N + n];
          if (RELU) val = fmaxf(val, 0.f);
          if (outf) outf[(size_t)m * N + n] = val;
          if (outh) outh[(size_t)m * N + n] = (f16)val;
        }
      }
    }
  }
}

// ---------------------------------------------------------------- LayerNorm -> f16
__global__ __launch_bounds__(256) void ln_kernel(
    const float* __restrict__ x, const float* __restrict__ g,
    const float* __restrict__ b, f16* __restrict__ out) {
  __shared__ float red[256];
  int tid = threadIdx.x;
  const float* row = x + (size_t)blockIdx.x * DMODEL;
  float v[4];
  float s = 0.f;
#pragma unroll
  for (int i = 0; i < 4; ++i) { v[i] = row[tid + i * 256]; s += v[i]; }
  red[tid] = s; __syncthreads();
  for (int o = 128; o > 0; o >>= 1) { if (tid < o) red[tid] += red[tid + o]; __syncthreads(); }
  float mean = red[0] * (1.f / DMODEL);
  __syncthreads();
  float vs = 0.f;
#pragma unroll
  for (int i = 0; i < 4; ++i) { float d = v[i] - mean; vs += d * d; }
  red[tid] = vs; __syncthreads();
  for (int o = 128; o > 0; o >>= 1) { if (tid < o) red[tid] += red[tid + o]; __syncthreads(); }
  float rstd = rsqrtf(red[0] * (1.f / DMODEL) + 1e-5f);
  f16* orow = out + (size_t)blockIdx.x * DMODEL;
#pragma unroll
  for (int i = 0; i < 4; ++i) {
    int d = tid + i * 256;
    float y = (v[i] - mean) * rstd;
    if (g) y = y * g[d] + b[d];
    orow[d] = (f16)y;
  }
}

// ---------------------------------------------------------------- embedding gather
__global__ __launch_bounds__(256) void embed_kernel(
    const int* __restrict__ src, const float* __restrict__ emb, float* __restrict__ x) {
  int row = blockIdx.x;                 // s*B + b
  int tok = src[row];
  const float* e = emb + (size_t)tok * DMODEL;
  float* xr = x + (size_t)row * DMODEL;
  for (int i = threadIdx.x; i < DMODEL; i += 256) xr[i] = e[i];
}

// ---------------------------------------------------------------- fp32 -> f16
__global__ __launch_bounds__(256) void cvt_kernel(
    const float* __restrict__ in, f16* __restrict__ out, int n) {
  int i = blockIdx.x * 256 + threadIdx.x;
  if (i < n) out[i] = (f16)in[i];
}

// ---------------------------------------------------------------- RoPE + head pack
// proj (M,3D) f16 -> q,k as (BH,S,HD) f16 (rope'd), v transposed as (BH,HD,S) f16
__global__ __launch_bounds__(256) void rope_kernel(
    const f16* __restrict__ proj, f16* __restrict__ q, f16* __restrict__ k,
    f16* __restrict__ vT) {
  int gid = blockIdx.x * 256 + threadIdx.x;   // BH*S*(HD/2) threads
  int j  = gid & 31;                          // rotary pair index, HD/2=32
  int s  = (gid >> 5) & (SEQ - 1);
  int bh = gid >> 15;                         // b*NHEAD + h
  int b_ = bh / NHEAD, h_ = bh % NHEAD;
  const f16* pr = proj + (size_t)(s * BATCH + b_) * (3 * DMODEL);
  int hd0 = h_ * HDIM + 2 * j;
  float q0 = (float)pr[hd0],              q1 = (float)pr[hd0 + 1];
  float k0 = (float)pr[DMODEL + hd0],     k1 = (float)pr[DMODEL + hd0 + 1];
  float v0 = (float)pr[2 * DMODEL + hd0], v1 = (float)pr[2 * DMODEL + hd0 + 1];
  float freq = __powf(10000.f, -(float)(2 * j) / (float)HDIM);
  float ang = (float)s * freq;
  float sn = __sinf(ang), cs = __cosf(ang);
  float qe = q0 * cs - q1 * sn, qo = q1 * cs + q0 * sn;
  float ke = k0 * cs - k1 * sn, ko = k1 * cs + k0 * sn;
  size_t qb = (size_t)bh * (SEQ * HDIM) + (size_t)s * HDIM + 2 * j;
  q[qb] = (f16)qe; q[qb + 1] = (f16)qo;
  k[qb] = (f16)ke; k[qb + 1] = (f16)ko;
  size_t vb = (size_t)bh * (HDIM * SEQ);
  vT[vb + (size_t)(2 * j) * SEQ + s]     = (f16)v0;
  vT[vb + (size_t)(2 * j + 1) * SEQ + s] = (f16)v1;
}

// ---------------------------------------------------------------- causal flash attention
// One wave per (bh, 16-query tile). QK^T and PV both via WMMA.
__global__ __launch_bounds__(32) void attn_kernel(
    const f16* __restrict__ q, const f16* __restrict__ k,
    const f16* __restrict__ vT, f16* __restrict__ out /* (M,D) f16 */) {
  __shared__ __align__(16) f16 Ptile[16 * 32];
  int qt = blockIdx.x;                   // query tile 0..S/16-1
  int bh = blockIdx.y;
  int lane = threadIdx.x;
  int b_ = bh / NHEAD, h_ = bh % NHEAD;
  const f16* qb = q  + (size_t)bh * (SEQ * HDIM);
  const f16* kb = k  + (size_t)bh * (SEQ * HDIM);
  const f16* vb = vT + (size_t)bh * (HDIM * SEQ);

  // Q fragments (16x64 split into two 16x32 A fragments), loaded once
  FragU qf[2];
  {
    int arow = qt * 16 + (lane & 15);
    int ko = (lane >> 4) << 4;
#pragma unroll
    for (int t = 0; t < 2; ++t) {
      const f16* p = qb + (size_t)arow * HDIM + t * 32 + ko;
#pragma unroll
      for (int i = 0; i < 4; ++i) {
        qf[t].u[i]     = *(const unsigned*)(p + 2 * i);
        qf[t].u[i + 4] = *(const unsigned*)(p + 8 + 2 * i);
      }
    }
  }

  v8f o[4] = {};
  float mst[8], lst[8];
#pragma unroll
  for (int r = 0; r < 8; ++r) { mst[r] = -1e30f; lst[r] = 0.f; }

  int rowm = qt * 16 + ((lane >> 4) << 3);   // + r = this lane's rows
  int kblocks = (qt * 16 + 15) / 32 + 1;     // causal 32-key blocks

  for (int kbI = 0; kbI < kblocks; ++kbI) {
    int ks = kbI * 32;
    v8f sc[2] = {};
    // scores: Q(16x64) @ K^T(64x16) twice (two 16-key subtiles)
#pragma unroll
    for (int c = 0; c < 2; ++c) {
      int key = ks + c * 16 + (lane & 15);
#pragma unroll
      for (int t = 0; t < 2; ++t) {
        FragU bf;
        int hdb = t * 32 + ((lane >> 4) << 4);
        const f16* p = kb + (size_t)key * HDIM + hdb;
#pragma unroll
        for (int i = 0; i < 8; ++i) bf.u[i] = *(const unsigned*)(p + 2 * i);
        sc[c] = __builtin_amdgcn_wmma_f32_16x16x32_f16(false, qf[t].v, false, bf.v,
                                                       (short)0, sc[c], false, false);
      }
    }
    // scale + causal mask
#pragma unroll
    for (int c = 0; c < 2; ++c) {
      int key = ks + c * 16 + (lane & 15);
#pragma unroll
      for (int r = 0; r < 8; ++r) {
        float v = sc[c][r] * 0.125f;        // 1/sqrt(64)
        if (key > rowm + r) v = -1e30f;
        sc[c][r] = v;
      }
    }
    // online softmax (row reductions across the 16 lanes holding one row)
    float corr[8];
#pragma unroll
    for (int r = 0; r < 8; ++r) {
      float mx = fmaxf(sc[0][r], sc[1][r]);
      for (int off = 1; off < 16; off <<= 1) mx = fmaxf(mx, __shfl_xor(mx, off, 32));
      float mnew = fmaxf(mst[r], mx);
      corr[r] = __expf(mst[r] - mnew);
      float p0 = __expf(sc[0][r] - mnew);
      float p1 = __expf(sc[1][r] - mnew);
      float rs = p0 + p1;
      for (int off = 1; off < 16; off <<= 1) rs += __shfl_xor(rs, off, 32);
      lst[r] = lst[r] * corr[r] + rs;
      mst[r] = mnew;
      sc[0][r] = p0; sc[1][r] = p1;
    }
#pragma unroll
    for (int g = 0; g < 4; ++g)
#pragma unroll
      for (int r = 0; r < 8; ++r) o[g][r] *= corr[r];

    // P: C-layout -> LDS -> A-layout fragment
#pragma unroll
    for (int c = 0; c < 2; ++c)
#pragma unroll
      for (int r = 0; r < 8; ++r)
        Ptile[(r + ((lane >> 4) << 3)) * 32 + c * 16 + (lane & 15)] = (f16)sc[c][r];
    __syncthreads();
    FragU pf;
    {
      int ko = (lane >> 4) << 4;
      const f16* p = Ptile + (lane & 15) * 32 + ko;
#pragma unroll
      for (int i = 0; i < 4; ++i) {
        pf.u[i]     = *(const unsigned*)(p + 2 * i);
        pf.u[i + 4] = *(const unsigned*)(p + 8 + 2 * i);
      }
    }
    __syncthreads();
    // O += P(16x32) @ V(32x64); V transposed so K pairs are contiguous
#pragma unroll
    for (int g = 0; g < 4; ++g) {
      FragU bf;
      int hd = g * 16 + (lane & 15);
      int kk = (lane >> 4) << 4;
      const f16* p = vb + (size_t)hd * SEQ + ks + kk;
#pragma unroll
      for (int i = 0; i < 8; ++i) bf.u[i] = *(const unsigned*)(p + 2 * i);
      o[g] = __builtin_amdgcn_wmma_f32_16x16x32_f16(false, pf.v, false, bf.v,
                                                    (short)0, o[g], false, false);
    }
  }

  // normalize + scatter back to (S,B,D) f16
#pragma unroll
  for (int g = 0; g < 4; ++g) {
    int hd = g * 16 + (lane & 15);
#pragma unroll
    for (int r = 0; r < 8; ++r) {
      int s = rowm + r;
      float val = o[g][r] / lst[r];
      out[(size_t)(s * BATCH + b_) * DMODEL + h_ * HDIM + hd] = (f16)val;
    }
  }
}

// ---------------------------------------------------------------- host orchestration
static inline void* carve(char*& p, size_t bytes) {
  void* r = p;
  p += (bytes + 255) & ~(size_t)255;
  return r;
}

extern "C" void kernel_launch(void* const* d_in, const int* in_sizes, int n_in,
                              void* d_out, int out_size, void* d_ws, size_t ws_size,
                              hipStream_t stream) {
  const int*   src  = (const int*)d_in[0];
  const float* emb  = (const float*)d_in[1];
  const float* wqkv = (const float*)d_in[2];
  const float* bqkv = (const float*)d_in[3];
  const float* wo   = (const float*)d_in[4];
  const float* bo   = (const float*)d_in[5];
  const float* g1   = (const float*)d_in[6];
  const float* b1   = (const float*)d_in[7];
  const float* g2   = (const float*)d_in[8];
  const float* b2   = (const float*)d_in[9];
  const float* w1   = (const float*)d_in[10];
  const float* bf1  = (const float*)d_in[11];
  const float* w2   = (const float*)d_in[12];
  const float* bf2  = (const float*)d_in[13];
  const float* wp   = (const float*)d_in[14];
  const float* bp   = (const float*)d_in[15];
  float* out = (float*)d_out;

  char* ws = (char*)d_ws;
  const int nWqkv = LNUM * 3 * DMODEL * DMODEL;
  const int nWo   = LNUM * DMODEL * DMODEL;
  const int nW1   = LNUM * FFDIM * DMODEL;
  const int nW2   = LNUM * DMODEL * FFDIM;
  const int nWp   = VOCAB * DMODEL;

  f16* wqkv_h = (f16*)carve(ws, (size_t)nWqkv * 2);
  f16* wo_h   = (f16*)carve(ws, (size_t)nWo * 2);
  f16* w1_h   = (f16*)carve(ws, (size_t)nW1 * 2);
  f16* w2_h   = (f16*)carve(ws, (size_t)nW2 * 2);
  f16* wp_h   = (f16*)carve(ws, (size_t)nWp * 2);
  float* x    = (float*)carve(ws, (size_t)MROWS * DMODEL * 4);
  f16* hln    = (f16*)carve(ws, (size_t)MROWS * DMODEL * 2);
  f16* projh  = (f16*)carve(ws, (size_t)MROWS * 3 * DMODEL * 2);
  f16* qh     = (f16*)carve(ws, (size_t)BATCH * NHEAD * SEQ * HDIM * 2);
  f16* kh     = (f16*)carve(ws, (size_t)BATCH * NHEAD * SEQ * HDIM * 2);
  f16* vth    = (f16*)carve(ws, (size_t)BATCH * NHEAD * HDIM * SEQ * 2);
  f16* attnh  = (f16*)carve(ws, (size_t)MROWS * DMODEL * 2);
  f16* h1     = (f16*)carve(ws, (size_t)MROWS * FFDIM * 2);

  // weight fp32 -> f16 staging
  cvt_kernel<<<(nWqkv + 255) / 256, 256, 0, stream>>>(wqkv, wqkv_h, nWqkv);
  cvt_kernel<<<(nWo   + 255) / 256, 256, 0, stream>>>(wo,   wo_h,   nWo);
  cvt_kernel<<<(nW1   + 255) / 256, 256, 0, stream>>>(w1,   w1_h,   nW1);
  cvt_kernel<<<(nW2   + 255) / 256, 256, 0, stream>>>(w2,   w2_h,   nW2);
  cvt_kernel<<<(nWp   + 255) / 256, 256, 0, stream>>>(wp,   wp_h,   nWp);

  embed_kernel<<<MROWS, 256, 0, stream>>>(src, emb, x);

  dim3 gQKV((3 * DMODEL) / 128, MROWS / 128);
  dim3 gD(DMODEL / 128, MROWS / 128);
  dim3 gFF(FFDIM / 128, MROWS / 128);
  dim3 gPred((VOCAB + 127) / 128, MROWS / 128);
  dim3 gAttn(SEQ / 16, BATCH * NHEAD);
  int ropeBlocks = (BATCH * NHEAD * SEQ * (HDIM / 2)) / 256;

  for (int l = 0; l < LNUM; ++l) {
    ln_kernel<<<MROWS, 256, 0, stream>>>(x, g1 + l * DMODEL, b1 + l * DMODEL, hln);
    gemm_kernel<false><<<gQKV, 256, 0, stream>>>(
        hln, wqkv_h + (size_t)l * 3 * DMODEL * DMODEL, bqkv + l * 3 * DMODEL,
        nullptr, nullptr, projh, MROWS, 3 * DMODEL, DMODEL);
    rope_kernel<<<ropeBlocks, 256, 0, stream>>>(projh, qh, kh, vth);
    attn_kernel<<<gAttn, 32, 0, stream>>>(qh, kh, vth, attnh);
    gemm_kernel<false><<<gD, 256, 0, stream>>>(
        attnh, wo_h + (size_t)l * DMODEL * DMODEL, bo + l * DMODEL,
        x, x, nullptr, MROWS, DMODEL, DMODEL);
    ln_kernel<<<MROWS, 256, 0, stream>>>(x, g2 + l * DMODEL, b2 + l * DMODEL, hln);
    gemm_kernel<true><<<gFF, 256, 0, stream>>>(
        hln, w1_h + (size_t)l * FFDIM * DMODEL, bf1 + l * FFDIM,
        nullptr, nullptr, h1, MROWS, FFDIM, DMODEL);
    gemm_kernel<false><<<gD, 256, 0, stream>>>(
        h1, w2_h + (size_t)l * DMODEL * FFDIM, bf2 + l * DMODEL,
        x, x, nullptr, MROWS, DMODEL, FFDIM);
  }

  ln_kernel<<<MROWS, 256, 0, stream>>>(x, nullptr, nullptr, hln);
  gemm_kernel<false><<<gPred, 256, 0, stream>>>(
      hln, wp_h, bp, nullptr, out, nullptr, MROWS, VOCAB, DMODEL);
}
